// MDMDenoiserROPE_44693429682432
// MI455X (gfx1250) — matrix operand order repaired and checked by reading, not compiled, not run on hardware
//
#include <hip/hip_runtime.h>
#include <hip/hip_bf16.h>
#include <math.h>

// ---------------- problem constants ----------------
#define B_   32
#define T_   512
#define DIN_ 150
#define D_   512
#define L_   8
#define H_   8
#define HD_  64
#define FF_  2048
#define M_   (B_ * T_)   // 16384 token rows
#define KIN_ 160         // DIN_ padded to multiple of 32

// ---------------- WMMA plumbing ----------------
typedef __attribute__((ext_vector_type(16))) __bf16 v16bf;
typedef __attribute__((ext_vector_type(8)))  float  v8f;

union Frag { v16bf v; uint4 q[2]; };
union Acc  { v8f v; float f[8]; };

__device__ __forceinline__ v8f wmma_bf16(v16bf a, v16bf b, v8f c) {
  // D = A(16x32 bf16) * B(32x16 bf16) + C(16x16 f32)
  return __builtin_amdgcn_wmma_f32_16x16x32_bf16(false, a, false, b, (short)0, c,
                                                 false, false);
}

// ---------------- async global->LDS copy (CDNA5), guarded ----------------
#if defined(__has_builtin)
#if __has_builtin(__builtin_amdgcn_global_load_async_to_lds_b128)
#define HAVE_ASYNC 1
#endif
#endif
#ifndef HAVE_ASYNC
#define HAVE_ASYNC 0
#endif

typedef int v4i_ __attribute__((vector_size(16)));

__device__ __forceinline__ void cp16(void* lds, const void* g) {
#if HAVE_ASYNC
  __builtin_amdgcn_global_load_async_to_lds_b128(
      (__attribute__((address_space(1))) v4i_*)(v4i_*)(void*)g,
      (__attribute__((address_space(3))) v4i_*)(v4i_*)lds, 0, 0);
#else
  *(uint4*)lds = *(const uint4*)g;
#endif
}

__device__ __forceinline__ void cp_wait() {
#if HAVE_ASYNC
#if __has_builtin(__builtin_amdgcn_s_wait_asynccnt)
  __builtin_amdgcn_s_wait_asynccnt(0);
#else
  asm volatile("s_wait_asynccnt 0x0" ::: "memory");
#endif
#endif
}

// ---------------- conditioning kernels ----------------
__global__ __launch_bounds__(256) void cond_pe_kernel(const int* __restrict__ ts,
                                                      float* __restrict__ t_pe) {
  int idx = blockIdx.x * 256 + threadIdx.x;              // B*D = 16384
  if (idx >= B_ * D_) return;
  int b = idx >> 9, dd = idx & 511;
  int p = dd >> 1;
  float div = __expf(-logf(10000.0f) * (float)(2 * p) / (float)D_);
  float ang = (float)ts[b] * div;
  t_pe[idx] = (dd & 1) ? cosf(ang) : sinf(ang);          // interleaved sin/cos
}

__global__ __launch_bounds__(256) void txt_mean_kernel(const float* __restrict__ enc,
                                                       float* __restrict__ txtm) {
  int idx = blockIdx.x * 256 + threadIdx.x;              // B*512
  if (idx >= B_ * 512) return;
  int b = idx >> 9, d = idx & 511;
  float s = 0.f;
  for (int t = 0; t < 20; ++t) s += enc[((size_t)b * 20 + t) * 512 + d];
  txtm[idx] = s * (1.0f / 20.0f);
}

__global__ __launch_bounds__(256) void cond_mlp1_kernel(const float* __restrict__ t_pe,
                                                        const float* __restrict__ W,
                                                        const float* __restrict__ bias,
                                                        float* __restrict__ outp) {
  int idx = blockIdx.x * 256 + threadIdx.x;              // B*D
  if (idx >= B_ * D_) return;
  int b = idx >> 9, n = idx & 511;
  float acc = bias[n];
  const float* row = t_pe + (size_t)b * D_;
  for (int k = 0; k < D_; ++k) acc += row[k] * W[(size_t)k * D_ + n];
  outp[idx] = acc / (1.0f + __expf(-acc));               // SiLU
}

__global__ __launch_bounds__(256) void cond_emb_kernel(const float* __restrict__ t_hid,
                                                       const float* __restrict__ W_t2,
                                                       const float* __restrict__ b_t2,
                                                       const float* __restrict__ txtm,
                                                       const float* __restrict__ W_txt,
                                                       const float* __restrict__ b_txt,
                                                       float* __restrict__ emb) {
  int idx = blockIdx.x * 256 + threadIdx.x;              // B*D
  if (idx >= B_ * D_) return;
  int b = idx >> 9, n = idx & 511;
  float acc = b_t2[n] + b_txt[n];
  const float* r1 = t_hid + (size_t)b * D_;
  const float* r2 = txtm + (size_t)b * 512;
  for (int k = 0; k < D_; ++k) acc += r1[k] * W_t2[(size_t)k * D_ + n];
  for (int k = 0; k < 512; ++k) acc += r2[k] * W_txt[(size_t)k * D_ + n];
  emb[idx] = acc;
}

__global__ __launch_bounds__(256) void rope_tab_kernel(float* __restrict__ rc,
                                                       float* __restrict__ rs) {
  int idx = blockIdx.x * 256 + threadIdx.x;              // T*32
  if (idx >= T_ * 32) return;
  int t = idx >> 5, d = idx & 31;
  float inv = __expf(-logf(10000.0f) * (float)d / 32.0f);
  float ang = (float)t * inv;
  rc[idx] = cosf(ang);
  rs[idx] = sinf(ang);
}

// ---------------- weight transpose+convert: [K,N] f32 -> [N,Kp] bf16 ----------------
__global__ __launch_bounds__(256) void wtrans_kernel(const float* __restrict__ W,
                                                     __hip_bfloat16* __restrict__ Wt,
                                                     int K, int N, int Kp, int total) {
  for (int idx = blockIdx.x * 256 + threadIdx.x; idx < total; idx += gridDim.x * 256) {
    int l = idx / (N * Kp);
    int r = idx - l * (N * Kp);
    int n = r / Kp, k = r - n * Kp;
    float v = (k < K) ? W[(size_t)l * K * N + (size_t)k * N + n] : 0.0f;
    Wt[idx] = __float2bfloat16(v);
  }
}

// ---------------- layernorm -> bf16 ----------------
__global__ __launch_bounds__(128) void layernorm_kernel(const float* __restrict__ hb,
                                                        const float* __restrict__ g,
                                                        const float* __restrict__ bta,
                                                        __hip_bfloat16* __restrict__ y) {
  __shared__ float s1[128];
  __shared__ float s2[128];
  int row = blockIdx.x, tid = threadIdx.x;
  const float4 xv = *(const float4*)(hb + (size_t)row * D_ + tid * 4);
  float sum = xv.x + xv.y + xv.z + xv.w;
  float sq = xv.x * xv.x + xv.y * xv.y + xv.z * xv.z + xv.w * xv.w;
  s1[tid] = sum; s2[tid] = sq;
  __syncthreads();
  for (int off = 64; off > 0; off >>= 1) {
    if (tid < off) { s1[tid] += s1[tid + off]; s2[tid] += s2[tid + off]; }
    __syncthreads();
  }
  float mean = s1[0] * (1.0f / D_);
  float var = s2[0] * (1.0f / D_) - mean * mean;
  float rstd = rsqrtf(var + 1e-5f);
  const float xe[4] = {xv.x, xv.y, xv.z, xv.w};
  for (int j = 0; j < 4; ++j) {
    int n = tid * 4 + j;
    y[(size_t)row * D_ + n] = __float2bfloat16((xe[j] - mean) * rstd * g[n] + bta[n]);
  }
}

// ---------------- generic WMMA GEMM (Wt pre-transposed bf16 [N][Kp]) ----------------
enum { EPI_BF16 = 0, EPI_RESID = 1, EPI_GELU = 2, EPI_F32 = 3, EPI_F32_EMB = 4 };

template <typename AT, int EPI>
__global__ __launch_bounds__(256) void gemm_wmma_kernel(
    const AT* __restrict__ A, const __hip_bfloat16* __restrict__ Wt,
    const float* __restrict__ bias, float* __restrict__ Cf,
    __hip_bfloat16* __restrict__ Cb, const float* __restrict__ emb,
    int M, int N, int K, int Kp) {
  constexpr int BM = 128, BN = 128, BK = 32, KP = 40;
  __shared__ alignas(16) __hip_bfloat16 As[BM][KP];
  __shared__ alignas(16) __hip_bfloat16 Bs[BN][KP];

  const int tid = threadIdx.x;
  const int lane = tid & 31, wave = tid >> 5;
  const int wm = wave >> 2, wn = wave & 3;               // 2 x 4 wave grid
  const int ln = lane & 15, lh = lane >> 4;
  const int m0 = blockIdx.y * BM;
  const int n0 = blockIdx.x * BN;

  Acc acc[4][2];
#pragma unroll
  for (int ms = 0; ms < 4; ++ms)
#pragma unroll
    for (int ns = 0; ns < 2; ++ns)
#pragma unroll
      for (int r = 0; r < 8; ++r) acc[ms][ns].f[r] = 0.0f;

  const int ksteps = Kp / BK;
  for (int kt = 0; kt < ksteps; ++kt) {
    const int k0 = kt * BK;
    __syncthreads();
    // ---- stage A ----
    if constexpr (sizeof(AT) == 2) {
      // bf16 activations: pure 16B async copies (K always multiple of 32 here)
#pragma unroll
      for (int j = 0; j < 2; ++j) {
        int e = tid + j * 256;                           // 512 chunks of 8 bf16
        int row = e >> 2, ch = e & 3;
        cp16(&As[row][ch * 8], A + (size_t)(m0 + row) * K + k0 + ch * 8);
      }
    } else {
      // fp32 activations: convert while staging; zero-pad k >= K
      const int ar = tid >> 1, ac = (tid & 1) * 16;
      const size_t arow = (size_t)(m0 + ar) * K;
      __builtin_prefetch((const void*)(A + arow + k0 + BK), 0, 1);
#pragma unroll
      for (int j = 0; j < 16; ++j) {
        int kk = k0 + ac + j;
        As[ar][ac + j] = (kk < K) ? __float2bfloat16(A[arow + kk])
                                  : __float2bfloat16(0.0f);
      }
    }
    // ---- stage B: Wt rows are K-contiguous bf16 ----
#pragma unroll
    for (int j = 0; j < 2; ++j) {
      int e = tid + j * 256;                             // 512 chunks of 8 bf16
      int row = e >> 2, ch = e & 3;
      int nn = n0 + row;
      if (nn < N) {
        cp16(&Bs[row][ch * 8], Wt + (size_t)nn * Kp + k0 + ch * 8);
      } else {
        uint4 z = {0u, 0u, 0u, 0u};
        *(uint4*)&Bs[row][ch * 8] = z;
      }
    }
    cp_wait();
    __syncthreads();
    // ---- fragment loads (ds_load_b128) + 8 WMMAs ----
    Frag a[4], bf[2];
#pragma unroll
    for (int ms = 0; ms < 4; ++ms) {
      const __hip_bfloat16* rp = &As[wm * 64 + ms * 16 + ln][lh * 8];
      a[ms].q[0] = *(const uint4*)rp;                    // K kbase..kbase+7
      a[ms].q[1] = *(const uint4*)(rp + 16);             // K 16+kbase..+7
    }
#pragma unroll
    for (int ns = 0; ns < 2; ++ns) {
      const __hip_bfloat16* rp = &Bs[wn * 32 + ns * 16 + ln][lh * 8];
      bf[ns].q[0] = *(const uint4*)rp;                   // K kbase..kbase+7
      bf[ns].q[1] = *(const uint4*)(rp + 16);            // K 16+kbase..+7
    }
#pragma unroll
    for (int ms = 0; ms < 4; ++ms)
#pragma unroll
      for (int ns = 0; ns < 2; ++ns)
        acc[ms][ns].v = wmma_bf16(a[ms].v, bf[ns].v, acc[ms][ns].v);
  }

  // ---- epilogue ----
#pragma unroll
  for (int ms = 0; ms < 4; ++ms) {
#pragma unroll
    for (int ns = 0; ns < 2; ++ns) {
      const int n = n0 + wn * 32 + ns * 16 + ln;
      if (n >= N) continue;
      const float bv = bias ? bias[n] : 0.0f;
#pragma unroll
      for (int r = 0; r < 8; ++r) {
        const int m = m0 + wm * 64 + ms * 16 + r + lh * 8;
        const float v = acc[ms][ns].f[r] + bv;
        const size_t idx = (size_t)m * N + n;
        if (EPI == EPI_BF16) {
          Cb[idx] = __float2bfloat16(v);
        } else if (EPI == EPI_GELU) {
          float gg = 0.5f * v * (1.0f + erff(v * 0.70710678f));
          Cb[idx] = __float2bfloat16(gg);
        } else if (EPI == EPI_RESID) {
          Cf[idx] += v;
        } else if (EPI == EPI_F32) {
          Cf[idx] = v;
        } else {  // EPI_F32_EMB: add per-batch conditioning (T = 512 -> m>>9)
          Cf[idx] = v + emb[(size_t)((m >> 9) * D_ + n)];
        }
      }
    }
  }
}

// ------- RoPE + repack qkv -> q,k [B,H,T,64] bf16 and v^T [B,H,64,T] bf16 -------
__global__ __launch_bounds__(256) void rope_repack_kernel(
    const __hip_bfloat16* __restrict__ qkv, const float* __restrict__ rc,
    const float* __restrict__ rs, __hip_bfloat16* __restrict__ qb,
    __hip_bfloat16* __restrict__ kb, __hip_bfloat16* __restrict__ vbT) {
  int idx = blockIdx.x * 256 + threadIdx.x;              // B*H*T*32
  int d = idx & 31;
  int t = (idx >> 5) & (T_ - 1);
  int hh = (idx >> 14) & (H_ - 1);
  int b = idx >> 17;
  size_t mrow = ((size_t)b * T_ + t) * (3 * D_);
  int c0 = hh * HD_ + d;
  float c = rc[t * 32 + d], s = rs[t * 32 + d];
  float q1 = __bfloat162float(qkv[mrow + c0]);
  float q2 = __bfloat162float(qkv[mrow + c0 + 32]);
  float k1 = __bfloat162float(qkv[mrow + D_ + c0]);
  float k2 = __bfloat162float(qkv[mrow + D_ + c0 + 32]);
  float v1 = __bfloat162float(qkv[mrow + 2 * D_ + c0]);
  float v2 = __bfloat162float(qkv[mrow + 2 * D_ + c0 + 32]);
  size_t orow = (((size_t)b * H_ + hh) * T_ + t) * HD_ + d;
  const float qs = 0.125f;  // 1/sqrt(HD) folded into q
  qb[orow]      = __float2bfloat16((q1 * c - q2 * s) * qs);
  qb[orow + 32] = __float2bfloat16((q2 * c + q1 * s) * qs);
  kb[orow]      = __float2bfloat16(k1 * c - k2 * s);
  kb[orow + 32] = __float2bfloat16(k2 * c + k1 * s);
  size_t vrow = (((size_t)b * H_ + hh) * HD_ + d) * T_ + t;  // transposed
  vbT[vrow]            = __float2bfloat16(v1);
  vbT[vrow + 32 * T_]  = __float2bfloat16(v2);
}

// ---------------- attention: per (b, h, 32-query tile) ----------------
__global__ __launch_bounds__(256) void attention_kernel(
    const __hip_bfloat16* __restrict__ qb, const __hip_bfloat16* __restrict__ kb,
    const __hip_bfloat16* __restrict__ vbT, __hip_bfloat16* __restrict__ o) {
  __shared__ alignas(16) __hip_bfloat16 Qs[32][72];
  __shared__ alignas(16) __hip_bfloat16 Ks[64][72];
  __shared__ alignas(16) __hip_bfloat16 Vt[64][72];
  __shared__ alignas(16) float Sf[32][520];
  __shared__ alignas(16) __hip_bfloat16 Pb[32][520];
  __shared__ float red[32][8];
  __shared__ float rowv[32];

  const int qt = blockIdx.x, hh = blockIdx.y, b = blockIdx.z;
  const int tid = threadIdx.x;
  const int lane = tid & 31, wave = tid >> 5;
  const int qh = wave >> 2, ksub = wave & 3;             // 2 (q) x 4 (key/d) wave grid
  const int ln = lane & 15, lh = lane >> 4;
  const size_t base = ((size_t)b * H_ + hh) * T_ * HD_;  // q/k row-major base
  const size_t vtb = ((size_t)b * H_ + hh) * HD_ * T_;   // v^T base

  // stage Q tile [32,64] : 256 chunks of 8 bf16
  {
    int row = tid >> 3, ch = tid & 7;
    cp16(&Qs[row][ch * 8], qb + base + (size_t)(qt * 32 + row) * HD_ + ch * 8);
  }

  // ---- pass 1: S = Q K^T ----
  for (int kt = 0; kt < 8; ++kt) {
    __syncthreads();
#pragma unroll
    for (int j = 0; j < 2; ++j) {                        // K tile [64,64]: 512 chunks
      int e = tid + j * 256;
      int row = e >> 3, ch = e & 7;
      cp16(&Ks[row][ch * 8], kb + base + (size_t)(kt * 64 + row) * HD_ + ch * 8);
    }
    cp_wait();
    __syncthreads();
    Acc acc;
#pragma unroll
    for (int r = 0; r < 8; ++r) acc.f[r] = 0.0f;
#pragma unroll
    for (int ds = 0; ds < 2; ++ds) {                     // K-dim = HD = 64
      Frag a, bf;
      const __hip_bfloat16* ap = &Qs[qh * 16 + ln][ds * 32 + lh * 8];
      a.q[0] = *(const uint4*)ap;
      a.q[1] = *(const uint4*)(ap + 16);
      const __hip_bfloat16* bp = &Ks[ksub * 16 + ln][ds * 32 + lh * 8];
      bf.q[0] = *(const uint4*)bp;
      bf.q[1] = *(const uint4*)(bp + 16);
      acc.v = wmma_bf16(a.v, bf.v, acc.v);
    }
#pragma unroll
    for (int r = 0; r < 8; ++r)
      Sf[qh * 16 + r + lh * 8][kt * 64 + ksub * 16 + ln] = acc.f[r];
  }
  __syncthreads();

  // ---- softmax over 512 keys per query row ----
  const int row = tid >> 3, sub = tid & 7;
  float mx = -1e30f;
  for (int c = sub * 64; c < sub * 64 + 64; ++c) mx = fmaxf(mx, Sf[row][c]);
  red[row][sub] = mx;
  __syncthreads();
  if (sub == 0) {
    float m2 = red[row][0];
    for (int j = 1; j < 8; ++j) m2 = fmaxf(m2, red[row][j]);
    rowv[row] = m2;
  }
  __syncthreads();
  const float rm = rowv[row];
  float lsum = 0.0f;
  for (int c = sub * 64; c < sub * 64 + 64; ++c) {
    float e = __expf(Sf[row][c] - rm);
    Sf[row][c] = e;
    lsum += e;
  }
  red[row][sub] = lsum;
  __syncthreads();
  if (sub == 0) {
    float s = 0.0f;
    for (int j = 0; j < 8; ++j) s += red[row][j];
    rowv[row] = 1.0f / s;
  }
  __syncthreads();
  const float rinv = rowv[row];
  for (int c = sub * 64; c < sub * 64 + 64; ++c)
    Pb[row][c] = __float2bfloat16(Sf[row][c] * rinv);

  // ---- pass 2: O = P V ----
  Acc oacc;
#pragma unroll
  for (int r = 0; r < 8; ++r) oacc.f[r] = 0.0f;
  const int dt = wave & 3;
  for (int kt = 0; kt < 8; ++kt) {
    __syncthreads();
#pragma unroll
    for (int j = 0; j < 2; ++j) {                        // V^T tile [64 d][64 keys]
      int e = tid + j * 256;
      int d = e >> 3, ch = e & 7;
      cp16(&Vt[d][ch * 8], vbT + vtb + (size_t)d * T_ + kt * 64 + ch * 8);
    }
    cp_wait();
    __syncthreads();
#pragma unroll
    for (int ks = 0; ks < 2; ++ks) {                     // 32 keys per WMMA
      Frag a, bf;
      const __hip_bfloat16* ap = &Pb[qh * 16 + ln][kt * 64 + ks * 32 + lh * 8];
      a.q[0] = *(const uint4*)ap;
      a.q[1] = *(const uint4*)(ap + 16);
      const __hip_bfloat16* bp = &Vt[dt * 16 + ln][ks * 32 + lh * 8];
      bf.q[0] = *(const uint4*)bp;
      bf.q[1] = *(const uint4*)(bp + 16);
      oacc.v = wmma_bf16(a.v, bf.v, oacc.v);
    }
  }
#pragma unroll
  for (int r = 0; r < 8; ++r) {
    int q = qh * 16 + r + lh * 8;
    int d = dt * 16 + ln;
    o[((size_t)b * T_ + qt * 32 + q) * D_ + hh * HD_ + d] = __float2bfloat16(oacc.f[r]);
  }
}

// ---------------- launch ----------------
extern "C" void kernel_launch(void* const* d_in, const int* in_sizes, int n_in,
                              void* d_out, int out_size, void* d_ws, size_t ws_size,
                              hipStream_t stream) {
  const float* x       = (const float*)d_in[0];
  const int*   tsteps  = (const int*)d_in[1];
  const float* enc_txt = (const float*)d_in[2];
  const float* W_in    = (const float*)d_in[3];
  const float* b_in    = (const float*)d_in[4];
  const float* W_t1    = (const float*)d_in[5];
  const float* b_t1    = (const float*)d_in[6];
  const float* W_t2    = (const float*)d_in[7];
  const float* b_t2    = (const float*)d_in[8];
  const float* W_txt   = (const float*)d_in[9];
  const float* b_txt   = (const float*)d_in[10];
  const float* ln1_g   = (const float*)d_in[11];
  const float* ln1_b   = (const float*)d_in[12];
  const float* Wqkv    = (const float*)d_in[13];
  const float* bqkv    = (const float*)d_in[14];
  const float* Wo      = (const float*)d_in[15];
  const float* bo      = (const float*)d_in[16];
  const float* ln2_g   = (const float*)d_in[17];
  const float* ln2_b   = (const float*)d_in[18];
  const float* W1      = (const float*)d_in[19];
  const float* b1      = (const float*)d_in[20];
  const float* W2      = (const float*)d_in[21];
  const float* b2      = (const float*)d_in[22];
  const float* W_out   = (const float*)d_in[23];
  const float* b_out   = (const float*)d_in[24];
  float* out = (float*)d_out;

  char* p = (char*)d_ws;
  auto alloc = [&](size_t bytes) {
    char* r = p;
    p += (bytes + 255) & ~(size_t)255;
    return (void*)r;
  };
  float* t_pe  = (float*)alloc((size_t)B_ * D_ * 4);
  float* t_hid = (float*)alloc((size_t)B_ * D_ * 4);
  float* txtm  = (float*)alloc((size_t)B_ * 512 * 4);
  float* emb   = (float*)alloc((size_t)B_ * D_ * 4);
  float* ropec = (float*)alloc((size_t)T_ * 32 * 4);
  float* ropes = (float*)alloc((size_t)T_ * 32 * 4);
  float* hbuf  = (float*)alloc((size_t)M_ * D_ * 4);
  __hip_bfloat16* ybuf = (__hip_bfloat16*)alloc((size_t)M_ * D_ * 2);
  __hip_bfloat16* qkv  = (__hip_bfloat16*)alloc((size_t)M_ * 3 * D_ * 2);
  __hip_bfloat16* qb   = (__hip_bfloat16*)alloc((size_t)M_ * D_ * 2);
  __hip_bfloat16* kbb  = (__hip_bfloat16*)alloc((size_t)M_ * D_ * 2);
  __hip_bfloat16* vbT  = (__hip_bfloat16*)alloc((size_t)M_ * D_ * 2);
  __hip_bfloat16* obuf = (__hip_bfloat16*)alloc((size_t)M_ * D_ * 2);
  __hip_bfloat16* gbuf = (__hip_bfloat16*)alloc((size_t)M_ * FF_ * 2);
  // pre-transposed bf16 weights [N][Kp] (per layer)
  __hip_bfloat16* Wt_in  = (__hip_bfloat16*)alloc((size_t)D_ * KIN_ * 2);
  __hip_bfloat16* Wt_qkv = (__hip_bfloat16*)alloc((size_t)L_ * 3 * D_ * D_ * 2);
  __hip_bfloat16* Wt_o   = (__hip_bfloat16*)alloc((size_t)L_ * D_ * D_ * 2);
  __hip_bfloat16* Wt_1   = (__hip_bfloat16*)alloc((size_t)L_ * FF_ * D_ * 2);
  __hip_bfloat16* Wt_2   = (__hip_bfloat16*)alloc((size_t)L_ * D_ * FF_ * 2);
  __hip_bfloat16* Wt_out = (__hip_bfloat16*)alloc((size_t)DIN_ * D_ * 2);

  // weight transpose/convert (L folded into total)
  wtrans_kernel<<<320, 256, 0, stream>>>(W_in, Wt_in, DIN_, D_, KIN_, D_ * KIN_);
  wtrans_kernel<<<4096, 256, 0, stream>>>(Wqkv, Wt_qkv, D_, 3 * D_, D_, L_ * 3 * D_ * D_);
  wtrans_kernel<<<2048, 256, 0, stream>>>(Wo, Wt_o, D_, D_, D_, L_ * D_ * D_);
  wtrans_kernel<<<4096, 256, 0, stream>>>(W1, Wt_1, D_, FF_, D_, L_ * FF_ * D_);
  wtrans_kernel<<<4096, 256, 0, stream>>>(W2, Wt_2, FF_, D_, FF_, L_ * D_ * FF_);
  wtrans_kernel<<<300, 256, 0, stream>>>(W_out, Wt_out, D_, DIN_, D_, DIN_ * D_);

  // conditioning + RoPE tables
  cond_pe_kernel<<<64, 256, 0, stream>>>(tsteps, t_pe);
  txt_mean_kernel<<<64, 256, 0, stream>>>(enc_txt, txtm);
  cond_mlp1_kernel<<<64, 256, 0, stream>>>(t_pe, W_t1, b_t1, t_hid);
  cond_emb_kernel<<<64, 256, 0, stream>>>(t_hid, W_t2, b_t2, txtm, W_txt, b_txt, emb);
  rope_tab_kernel<<<64, 256, 0, stream>>>(ropec, ropes);

  // input projection + conditioning broadcast: h = x @ W_in + b_in + emb[b]
  gemm_wmma_kernel<float, EPI_F32_EMB><<<dim3(4, 128), 256, 0, stream>>>(
      x, Wt_in, b_in, hbuf, nullptr, emb, M_, D_, DIN_, KIN_);

  for (int i = 0; i < L_; ++i) {
    layernorm_kernel<<<M_, 128, 0, stream>>>(hbuf, ln1_g + i * D_, ln1_b + i * D_, ybuf);
    gemm_wmma_kernel<__hip_bfloat16, EPI_BF16><<<dim3(12, 128), 256, 0, stream>>>(
        ybuf, Wt_qkv + (size_t)i * 3 * D_ * D_, bqkv + i * 3 * D_, nullptr, qkv, nullptr,
        M_, 3 * D_, D_, D_);
    rope_repack_kernel<<<16384, 256, 0, stream>>>(qkv, ropec, ropes, qb, kbb, vbT);
    attention_kernel<<<dim3(16, H_, B_), 256, 0, stream>>>(qb, kbb, vbT, obuf);
    gemm_wmma_kernel<__hip_bfloat16, EPI_RESID><<<dim3(4, 128), 256, 0, stream>>>(
        obuf, Wt_o + (size_t)i * D_ * D_, bo + i * D_, hbuf, nullptr, nullptr,
        M_, D_, D_, D_);
    layernorm_kernel<<<M_, 128, 0, stream>>>(hbuf, ln2_g + i * D_, ln2_b + i * D_, ybuf);
    gemm_wmma_kernel<__hip_bfloat16, EPI_GELU><<<dim3(16, 128), 256, 0, stream>>>(
        ybuf, Wt_1 + (size_t)i * FF_ * D_, b1 + i * FF_, nullptr, gbuf, nullptr,
        M_, FF_, D_, D_);
    gemm_wmma_kernel<__hip_bfloat16, EPI_RESID><<<dim3(4, 128), 256, 0, stream>>>(
        gbuf, Wt_2 + (size_t)i * D_ * FF_, b2 + i * D_, hbuf, nullptr, nullptr,
        M_, D_, FF_, FF_);
  }

  // output projection (N = 150, guarded)
  gemm_wmma_kernel<float, EPI_F32><<<dim3(2, 128), 256, 0, stream>>>(
      hbuf, Wt_out, b_out, out, nullptr, nullptr, M_, DIN_, D_, D_);

  (void)in_sizes; (void)n_in; (void)out_size; (void)ws_size;
}
